// GNNReasoner_42030549958855
// MI455X (gfx1250) — matrix operand order repaired and checked by reading, not compiled, not run on hardware
//
#include <hip/hip_runtime.h>

#define NN 50000
#define NE 800000
#define FD 256   // IN_DIM = HEADS*HID = OUT_DIM = 256

typedef __attribute__((ext_vector_type(16))) __bf16 v16bf;
typedef __attribute__((ext_vector_type(8)))  float  v8f;

static inline int cdiv(int a, int b) { return (a + b - 1) / b; }

// ---------------- helpers ----------------
__device__ __forceinline__ unsigned short f2bf(float f) {
  unsigned int u = __float_as_uint(f);
  u += 0x7FFFu + ((u >> 16) & 1u);      // round-to-nearest-even
  return (unsigned short)(u >> 16);
}

__device__ __forceinline__ void atomicMaxF(float* addr, float v) {
  // monotonic-bits trick; init value must be -inf (or NaN bits)
  if (v >= 0.0f) atomicMax((int*)addr, __float_as_int(v));
  else           atomicMin((unsigned int*)addr, __float_as_uint(v));
}

// ---------------- conversion kernels ----------------
__global__ __launch_bounds__(256) void f32_to_bf16_kernel(const float* __restrict__ x,
                                                          unsigned short* __restrict__ y, int n) {
  int i = blockIdx.x * blockDim.x + threadIdx.x;
  if (i < n) y[i] = f2bf(x[i]);
}

// W[k*256+n] (row-major [K=256, N=256]) -> WT[n*256+k] bf16
__global__ __launch_bounds__(256) void conv_transpose_kernel(const float* __restrict__ W,
                                                             unsigned short* __restrict__ WT) {
  int i = blockIdx.x * blockDim.x + threadIdx.x;  // i < 65536
  int k = i >> 8, n = i & 255;
  WT[n * 256 + k] = f2bf(W[i]);
}

__global__ __launch_bounds__(256) void fill_kernel(float* __restrict__ p, float v, int n) {
  int i = blockIdx.x * blockDim.x + threadIdx.x;
  if (i < n) p[i] = v;
}

// ---------------- WMMA GEMM: C[M,256] = A[M,256] * B[256,256] ----------------
// A: bf16 row-major. BT: bf16, B transposed (BT[n*256+k]). C: f32 row-major.
// Block = 256 threads = 8 waves. Each wave: one 16-row tile x 64 cols (4 WMMA col-tiles).
__global__ __launch_bounds__(256) void gemm_bf16_wmma_kernel(const unsigned short* __restrict__ A,
                                                             const unsigned short* __restrict__ BT,
                                                             float* __restrict__ C, int M) {
  const int lane = threadIdx.x & 31;
  const int wave = threadIdx.x >> 5;
  const int tileM = blockIdx.x * 8 + wave;
  const int m0 = tileM * 16;
  if (m0 >= M) return;
  const int n0 = blockIdx.y * 64;
  const int half = lane >> 4;     // lane group (0: lanes 0-15, 1: lanes 16-31)
  const int l16  = lane & 15;
  const int rowA = m0 + l16;

  v8f acc[4] = {};
  for (int k0 = 0; k0 < 256; k0 += 32) {
    // A fragment, 16-bit 16x32 layout (ISA 7.12.2):
    //   half=0: VGPR0-3 = K k0..k0+7, VGPR4-7 = K k0+16..k0+23
    //   half=1: VGPR0-3 = K k0+8..k0+15, VGPR4-7 = K k0+24..k0+31
    union { v16bf v; uint4 u[2]; } a;
    const unsigned short* ap = A + (size_t)rowA * 256 + k0 + half * 8;
    a.u[0] = *(const uint4*)(ap);
    a.u[1] = *(const uint4*)(ap + 16);
#pragma unroll
    for (int nt = 0; nt < 4; ++nt) {
      // B fragment 32x16: lane holds column n = n0+nt*16+l16,
      //   half=0 -> K k0..k0+7, k0+16..k0+23; half=1 -> K k0+8..k0+15, k0+24..k0+31
      union { v16bf v; uint4 u[2]; } b;
      const unsigned short* bp = BT + (size_t)(n0 + nt * 16 + l16) * 256 + k0 + half * 8;
      b.u[0] = *(const uint4*)(bp);
      b.u[1] = *(const uint4*)(bp + 16);
      acc[nt] = __builtin_amdgcn_wmma_f32_16x16x32_bf16(
          false, a.v, false, b.v, (short)0, acc[nt], false, false);
    }
  }
  // C layout: VGPR r -> row m0 + half*8 + r, col = n-tile + l16
#pragma unroll
  for (int nt = 0; nt < 4; ++nt) {
#pragma unroll
    for (int r = 0; r < 8; ++r) {
      int row = m0 + half * 8 + r;
      C[(size_t)row * 256 + n0 + nt * 16 + l16] = acc[nt][r];
    }
  }
}

// ---------------- per-node attention coefficients el/er ----------------
// wave per node; 8 channels/lane; segmented reduction (width 16 for 2 heads, 32 for 1)
template <int HEADS>
__global__ __launch_bounds__(256) void attn_coef_kernel(const float* __restrict__ z,
                                                        const float* __restrict__ al,
                                                        const float* __restrict__ ar,
                                                        float* __restrict__ el,
                                                        float* __restrict__ er, int nN) {
  int node = blockIdx.x * 8 + (threadIdx.x >> 5);
  if (node >= nN) return;
  int lane = threadIdx.x & 31;
  int c0 = lane * 8;
  const float* zp = z + (size_t)node * 256 + c0;
  float sl = 0.f, sr = 0.f;
#pragma unroll
  for (int i = 0; i < 8; ++i) { float zv = zp[i]; sl += zv * al[c0 + i]; sr += zv * ar[c0 + i]; }
  const int W = (HEADS == 2) ? 16 : 32;
#pragma unroll
  for (int off = W / 2; off >= 1; off >>= 1) {
    sl += __shfl_down(sl, off, W);
    sr += __shfl_down(sr, off, W);
  }
  if ((lane & (W - 1)) == 0) {
    int h = lane / W;
    el[node * HEADS + h] = sl;
    er[node * HEADS + h] = sr;
  }
}

// ---------------- edge passes ----------------
template <int HEADS>
__global__ __launch_bounds__(256) void edge_max_kernel(const int* __restrict__ src,
                                                       const int* __restrict__ dst,
                                                       const float* __restrict__ el,
                                                       const float* __restrict__ er,
                                                       float* __restrict__ m, int nE) {
  int e = blockIdx.x * blockDim.x + threadIdx.x;
  if (e >= nE) return;
  int s = src[e], d = dst[e];
#pragma unroll
  for (int h = 0; h < HEADS; ++h) {
    float v = el[s * HEADS + h] + er[d * HEADS + h];
    v = v > 0.f ? v : 0.2f * v;                 // LeakyReLU(0.2)
    atomicMaxF(&m[d * HEADS + h], v);
  }
}

template <int HEADS>
__global__ __launch_bounds__(256) void edge_sum_kernel(const int* __restrict__ src,
                                                       const int* __restrict__ dst,
                                                       const float* __restrict__ el,
                                                       const float* __restrict__ er,
                                                       const float* __restrict__ m,
                                                       float* __restrict__ s, int nE) {
  int e = blockIdx.x * blockDim.x + threadIdx.x;
  if (e >= nE) return;
  int sn = src[e], dn = dst[e];
#pragma unroll
  for (int h = 0; h < HEADS; ++h) {
    float v = el[sn * HEADS + h] + er[dn * HEADS + h];
    v = v > 0.f ? v : 0.2f * v;
    atomicAdd(&s[dn * HEADS + h], expf(v - m[dn * HEADS + h]));
  }
}

// wave per edge: 2x float4 gather of z[src], 8 fp32 atomics into agg[dst]
template <int HEADS>
__global__ __launch_bounds__(256) void edge_agg_kernel(const int* __restrict__ src,
                                                       const int* __restrict__ dst,
                                                       const float* __restrict__ el,
                                                       const float* __restrict__ er,
                                                       const float* __restrict__ m,
                                                       const float* __restrict__ s,
                                                       const float* __restrict__ z,
                                                       float* __restrict__ agg, int nE) {
  int e = blockIdx.x * 8 + (threadIdx.x >> 5);
  if (e >= nE) return;
  int lane = threadIdx.x & 31;
  int sn = src[e], dn = dst[e];
  float alpha[HEADS];
#pragma unroll
  for (int h = 0; h < HEADS; ++h) {
    float v = el[sn * HEADS + h] + er[dn * HEADS + h];
    v = v > 0.f ? v : 0.2f * v;
    alpha[h] = expf(v - m[dn * HEADS + h]) / s[dn * HEADS + h];
  }
  int c0 = lane * 8;                                   // head boundary at 128; c0 never straddles
  float a1 = (HEADS == 2) ? alpha[HEADS - 1] : alpha[0];
  float al_ = ((HEADS == 2) && (c0 & 128)) ? a1 : alpha[0];
  const float4* zp = (const float4*)(z + (size_t)sn * 256 + c0);
  float4 za = zp[0], zb = zp[1];
  float* op = agg + (size_t)dn * 256 + c0;
  atomicAdd(op + 0, za.x * al_); atomicAdd(op + 1, za.y * al_);
  atomicAdd(op + 2, za.z * al_); atomicAdd(op + 3, za.w * al_);
  atomicAdd(op + 4, zb.x * al_); atomicAdd(op + 5, zb.y * al_);
  atomicAdd(op + 6, zb.z * al_); atomicAdd(op + 7, zb.w * al_);
}

// ---------------- bias (+optional relu) epilogue ----------------
__global__ __launch_bounds__(256) void bias_act_kernel(const float* __restrict__ in,
                                                       const float* __restrict__ b,
                                                       float* __restrict__ out, int n, int relu) {
  int i = blockIdx.x * blockDim.x + threadIdx.x;
  if (i >= n) return;
  float v = in[i] + b[i & 255];
  if (relu) v = v > 0.f ? v : 0.f;
  out[i] = v;
}

// ---------------- host launcher ----------------
extern "C" void kernel_launch(void* const* d_in, const int* in_sizes, int n_in,
                              void* d_out, int out_size, void* d_ws, size_t ws_size,
                              hipStream_t stream) {
  const float* x   = (const float*)d_in[0];
  const int*   src = (const int*)d_in[1];
  const int*   dst = (const int*)d_in[2];
  const float* W1  = (const float*)d_in[3];
  const float* al1 = (const float*)d_in[4];
  const float* ar1 = (const float*)d_in[5];
  const float* b1  = (const float*)d_in[6];
  const float* W2  = (const float*)d_in[7];
  const float* al2 = (const float*)d_in[8];
  const float* ar2 = (const float*)d_in[9];
  const float* b2  = (const float*)d_in[10];
  float* out = (float*)d_out;

  // workspace layout
  char* ws = (char*)d_ws;
  size_t o = 0;
  float* zbuf   = (float*)(ws + o); o += (size_t)NN * 256 * 4;   // z1, reused as z2
  float* aggbuf = (float*)(ws + o); o += (size_t)NN * 256 * 4;   // agg1 -> h (in place)
  unsigned short* featb = (unsigned short*)(ws + o); o += (size_t)NN * 256 * 2;  // xb -> hb
  unsigned short* W1T   = (unsigned short*)(ws + o); o += 65536 * 2;
  unsigned short* W2T   = (unsigned short*)(ws + o); o += 65536 * 2;
  float* el1 = (float*)(ws + o); o += NN * 2 * 4;
  float* er1 = (float*)(ws + o); o += NN * 2 * 4;
  float* m1  = (float*)(ws + o); o += NN * 2 * 4;
  float* s1  = (float*)(ws + o); o += NN * 2 * 4;
  float* el2 = (float*)(ws + o); o += NN * 4;
  float* er2 = (float*)(ws + o); o += NN * 4;
  float* m2  = (float*)(ws + o); o += NN * 4;
  float* s2  = (float*)(ws + o); o += NN * 4;

  const int T = 256;
  dim3 gG(cdiv(NN, 128), 4);   // GEMM grid: 8 waves x 16 rows = 128 rows/block; 4 x 64 cols

  // ---- prep ----
  f32_to_bf16_kernel<<<cdiv(NN * 256, T), T, 0, stream>>>(x, featb, NN * 256);
  conv_transpose_kernel<<<cdiv(65536, T), T, 0, stream>>>(W1, W1T);
  conv_transpose_kernel<<<cdiv(65536, T), T, 0, stream>>>(W2, W2T);
  fill_kernel<<<cdiv(NN * 2, T), T, 0, stream>>>(m1, -INFINITY, NN * 2);
  fill_kernel<<<cdiv(NN * 2, T), T, 0, stream>>>(s1, 0.f, NN * 2);
  fill_kernel<<<cdiv(NN * 256, T), T, 0, stream>>>(aggbuf, 0.f, NN * 256);
  fill_kernel<<<cdiv(NN, T), T, 0, stream>>>(m2, -INFINITY, NN);
  fill_kernel<<<cdiv(NN, T), T, 0, stream>>>(s2, 0.f, NN);
  fill_kernel<<<cdiv(NN * 256, T), T, 0, stream>>>(out, 0.f, NN * 256);

  // ---- layer 1 (2 heads x 128) ----
  gemm_bf16_wmma_kernel<<<gG, T, 0, stream>>>(featb, W1T, zbuf, NN);
  attn_coef_kernel<2><<<cdiv(NN, 8), T, 0, stream>>>(zbuf, al1, ar1, el1, er1, NN);
  edge_max_kernel<2><<<cdiv(NE, T), T, 0, stream>>>(src, dst, el1, er1, m1, NE);
  edge_sum_kernel<2><<<cdiv(NE, T), T, 0, stream>>>(src, dst, el1, er1, m1, s1, NE);
  edge_agg_kernel<2><<<cdiv(NE, 8), T, 0, stream>>>(src, dst, el1, er1, m1, s1, zbuf, aggbuf, NE);
  bias_act_kernel<<<cdiv(NN * 256, T), T, 0, stream>>>(aggbuf, b1, aggbuf, NN * 256, 1);

  // ---- layer 2 (1 head x 256) ----
  f32_to_bf16_kernel<<<cdiv(NN * 256, T), T, 0, stream>>>(aggbuf, featb, NN * 256);
  gemm_bf16_wmma_kernel<<<gG, T, 0, stream>>>(featb, W2T, zbuf, NN);
  attn_coef_kernel<1><<<cdiv(NN, 8), T, 0, stream>>>(zbuf, al2, ar2, el2, er2, NN);
  edge_max_kernel<1><<<cdiv(NE, T), T, 0, stream>>>(src, dst, el2, er2, m2, NE);
  edge_sum_kernel<1><<<cdiv(NE, T), T, 0, stream>>>(src, dst, el2, er2, m2, s2, NE);
  edge_agg_kernel<1><<<cdiv(NE, 8), T, 0, stream>>>(src, dst, el2, er2, m2, s2, zbuf, out, NE);
  bias_act_kernel<<<cdiv(NN * 256, T), T, 0, stream>>>(out, b2, out, NN * 256, 0);
}
